// Score_encoding_layer_12953621364793
// MI455X (gfx1250) — matrix operand order repaired and checked by reading, not compiled, not run on hardware
//
#include <hip/hip_runtime.h>
#include <hip/hip_bf16.h>
#include <math.h>

typedef __bf16 bf16_t;
typedef __attribute__((ext_vector_type(16))) __bf16 v16bf;
typedef __attribute__((ext_vector_type(8)))  __bf16 v8bf;
typedef __attribute__((ext_vector_type(4)))  __bf16 v4bf;
typedef __attribute__((ext_vector_type(8)))  float  v8f;
typedef __attribute__((ext_vector_type(4)))  float  f4;

#define NB 16
#define NQ 512
#define NP 4096
#define ND 256

union V16U { v16bf v; v8bf h[2]; };

// ---------------- conversion kernels ----------------
__global__ void k_conv_q(const float* __restrict__ q, bf16_t* __restrict__ qbf,
                         bf16_t* __restrict__ qT) {
  int i = blockIdx.x * blockDim.x + threadIdx.x;  // over NB*NQ*ND
  float v = q[i];
  bf16_t bv = (bf16_t)v;
  qbf[i] = bv;
  int d  = i % ND;
  int qq = (i / ND) % NQ;
  int b  = i / (ND * NQ);
  qT[(b * ND + d) * NQ + qq] = bv;
}

__global__ void k_conv_p(const float* __restrict__ p, bf16_t* __restrict__ pbf) {
  int i = blockIdx.x * blockDim.x + threadIdx.x;  // over NB*NP*ND
  pbf[i] = (bf16_t)p[i];
}

// ---------------- main attention kernel ----------------
// 128 threads = 4 waves; each wave owns 16 passage rows. Block covers 64 rows.
__launch_bounds__(128, 1)
__global__ void k_bidaf(const bf16_t* __restrict__ qbf, const bf16_t* __restrict__ qT,
                        const bf16_t* __restrict__ pbf, const float* __restrict__ pass,
                        float* __restrict__ out, float* __restrict__ max_sim) {
  __shared__ float  S[4][16][NQ];      // 128 KB: raw scores per wave
  __shared__ bf16_t Pb[4][16][NQ];     // 64 KB: exp(s - m) as bf16 (A-matrix source)
  __shared__ float  Lsum[4][16];

  const int b    = blockIdx.y;
  const int p0   = blockIdx.x * 64;
  const int wv   = threadIdx.x >> 5;
  const int ln   = threadIdx.x & 31;
  const int l15  = ln & 15;
  const int hl   = ln >> 4;            // half-wave select
  const int row0 = p0 + wv * 16;

  // Prefetch the f32 passage rows we will re-read in the output phase.
  __builtin_prefetch(pass + ((size_t)(b * NP + row0 + l15)) * ND + hl * 128, 0, 1);

  // ---- Resident A tile: passage rows (16 x 256 bf16), ISA A-layout ----
  V16U A[8];
  {
    const bf16_t* prow = pbf + ((size_t)(b * NP + row0 + l15)) * ND;
    #pragma unroll
    for (int c = 0; c < 8; ++c) {
      int kb = c * 32 + hl * 8;
      A[c].h[0] = *reinterpret_cast<const v8bf*>(prow + kb);
      A[c].h[1] = *reinterpret_cast<const v8bf*>(prow + kb + 16);
    }
  }

  // ---- Phase 1: scores = passage @ question^T, spill to LDS ----
  for (int qt = 0; qt < 32; ++qt) {
    v8f acc = {};
    const bf16_t* qb = qbf + ((size_t)(b * NQ + qt * 16 + l15)) * ND;
    #pragma unroll
    for (int c = 0; c < 8; ++c) {
      V16U Bm;
      int kb = c * 32 + hl * 8;        // B-layout: lane = column, K interleaved by 8 (matches A)
      Bm.h[0] = *reinterpret_cast<const v8bf*>(qb + kb);
      Bm.h[1] = *reinterpret_cast<const v8bf*>(qb + kb + 16);
      acc = __builtin_amdgcn_wmma_f32_16x16x32_bf16(false, A[c].v, false, Bm.v,
                                                    (short)0, acc, false, false);
    }
    #pragma unroll
    for (int v = 0; v < 8; ++v)        // C-layout: M = v + 8*hl, N = l15
      S[wv][hl * 8 + v][qt * 16 + l15] = acc[v];
  }
  __syncthreads();

  // ---- Phase 2: softmax over Q (two lanes per row) ----
  float m = -3.0e38f;
  {
    const float* srow = &S[wv][l15][hl * 256];
    #pragma unroll 4
    for (int i = 0; i < 64; ++i) {
      f4 s = *reinterpret_cast<const f4*>(srow + i * 4);
      m = fmaxf(m, fmaxf(fmaxf(s.x, s.y), fmaxf(s.z, s.w)));
    }
  }
  m = fmaxf(m, __shfl_xor(m, 16));
  if (hl == 0) max_sim[b * NP + row0 + l15] = m;

  float l = 0.f;
  {
    const float* srow = &S[wv][l15][hl * 256];
    bf16_t* prow = &Pb[wv][l15][hl * 256];
    for (int i = 0; i < 64; ++i) {
      f4 s = *reinterpret_cast<const f4*>(srow + i * 4);
      float e0 = __expf(s.x - m), e1 = __expf(s.y - m);
      float e2 = __expf(s.z - m), e3 = __expf(s.w - m);
      l += (e0 + e1) + (e2 + e3);
      v4bf pv = { (bf16_t)e0, (bf16_t)e1, (bf16_t)e2, (bf16_t)e3 };
      *reinterpret_cast<v4bf*>(prow + i * 4) = pv;
    }
  }
  l += __shfl_xor(l, 16);
  if (hl == 0) Lsum[wv][l15] = l;
  __syncthreads();

  // ---- Phase 3: c2q = P @ question (K = 512), plus outputs ----
  float rl[8];
  #pragma unroll
  for (int v = 0; v < 8; ++v) rl[v] = 1.0f / Lsum[wv][hl * 8 + v];

  V16U Aq[16];                          // resident P-weights, A-layout from LDS
  #pragma unroll
  for (int c = 0; c < 16; ++c) {
    int kb = c * 32 + hl * 8;
    Aq[c].h[0] = *reinterpret_cast<const v8bf*>(&Pb[wv][l15][kb]);
    Aq[c].h[1] = *reinterpret_cast<const v8bf*>(&Pb[wv][l15][kb + 16]);
  }

  for (int t = 0; t < 16; ++t) {
    v8f acc = {};
    const bf16_t* qtrow = qT + ((size_t)(b * ND + t * 16 + l15)) * NQ;
    #pragma unroll
    for (int c = 0; c < 16; ++c) {
      V16U Bm;
      int kb = c * 32 + hl * 8;
      Bm.h[0] = *reinterpret_cast<const v8bf*>(qtrow + kb);
      Bm.h[1] = *reinterpret_cast<const v8bf*>(qtrow + kb + 16);
      acc = __builtin_amdgcn_wmma_f32_16x16x32_bf16(false, Aq[c].v, false, Bm.v,
                                                    (short)0, acc, false, false);
    }
    int d = t * 16 + l15;
    #pragma unroll
    for (int v = 0; v < 8; ++v) {
      int p = row0 + hl * 8 + v;
      float c2q = acc[v] * rl[v];
      size_t ob = ((size_t)(b * NP + p)) * (4 * ND);
      float pf = pass[((size_t)(b * NP + p)) * ND + d];
      out[ob + ND + d]     = c2q;        // section 1: c2q
      out[ob + 2 * ND + d] = pf * c2q;   // section 2: passage * c2q
    }
  }
}

// ---------------- q2c statistics (max + sum-exp per batch) ----------------
__global__ void k_q2c_stats(const float* __restrict__ max_sim, float* __restrict__ MZ,
                            float* __restrict__ q2c) {
  __shared__ float red[256];
  __shared__ float Mb;
  int b = blockIdx.x, tid = threadIdx.x;
  const float* ms = max_sim + b * NP;
  float m = -3.0e38f;
  for (int i = tid; i < NP; i += 256) m = fmaxf(m, ms[i]);
  red[tid] = m; __syncthreads();
  for (int s = 128; s > 0; s >>= 1) {
    if (tid < s) red[tid] = fmaxf(red[tid], red[tid + s]);
    __syncthreads();
  }
  if (tid == 0) Mb = red[0];
  __syncthreads();
  float M = Mb;
  float z = 0.f;
  for (int i = tid; i < NP; i += 256) z += __expf(ms[i] - M);
  __syncthreads();
  red[tid] = z; __syncthreads();
  for (int s = 128; s > 0; s >>= 1) {
    if (tid < s) red[tid] += red[tid + s];
    __syncthreads();
  }
  if (tid == 0) { MZ[2 * b] = M; MZ[2 * b + 1] = red[0]; }
  q2c[b * ND + tid] = 0.f;              // zero the accumulator every launch
}

// ---------------- q2c accumulation (chunked, f32 atomics) ----------------
__global__ void k_q2c_acc(const float* __restrict__ max_sim, const float* __restrict__ MZ,
                          const float* __restrict__ pass, float* __restrict__ q2c) {
  __shared__ float w[256];
  int b = blockIdx.y, ch = blockIdx.x, tid = threadIdx.x;
  float M = MZ[2 * b], invZ = 1.0f / MZ[2 * b + 1];
  int p = ch * 256 + tid;
  w[tid] = __expf(max_sim[b * NP + p] - M) * invZ;
  __syncthreads();
  float acc = 0.f;
  const float* pr = pass + ((size_t)b * NP + ch * 256) * ND + tid;
  for (int j = 0; j < 256; ++j) acc += w[j] * pr[(size_t)j * ND];
  atomicAdd(&q2c[b * ND + tid], acc);
}

// ---------------- final merge: sections 0 and 3 ----------------
__global__ void k_merge(const float* __restrict__ pass, const float* __restrict__ q2c,
                        float* __restrict__ out) {
  size_t g = (size_t)blockIdx.x * blockDim.x + threadIdx.x;  // over NB*NP*ND/4
  int d4   = (int)(g & 63);
  size_t pp = g >> 6;
  int p = (int)(pp & (NP - 1));
  int b = (int)(pp >> 12);
  int d = d4 * 4;
  f4 pv = *reinterpret_cast<const f4*>(pass + ((size_t)(b * NP + p)) * ND + d);
  f4 qv = *reinterpret_cast<const f4*>(q2c + b * ND + d);
  size_t ob = ((size_t)(b * NP + p)) * (4 * ND);
  *reinterpret_cast<f4*>(out + ob + d) = pv;                 // section 0: passage
  f4 r = pv * qv;
  *reinterpret_cast<f4*>(out + ob + 3 * ND + d) = r;         // section 3: passage * q2c
}

extern "C" void kernel_launch(void* const* d_in, const int* in_sizes, int n_in,
                              void* d_out, int out_size, void* d_ws, size_t ws_size,
                              hipStream_t stream) {
  const float* q    = (const float*)d_in[0];  // [16,512,256]
  const float* pass = (const float*)d_in[1];  // [16,4096,256]
  float* out = (float*)d_out;                 // [16,4096,1024]

  char* ws = (char*)d_ws;
  bf16_t* qbf = (bf16_t*)ws;                                   // 4 MB
  bf16_t* qT  = (bf16_t*)(ws + (size_t)4  * 1024 * 1024);      // 4 MB
  bf16_t* pbf = (bf16_t*)(ws + (size_t)8  * 1024 * 1024);      // 32 MB
  float* max_sim = (float*)(ws + (size_t)40 * 1024 * 1024);    // 256 KB
  float* MZ      = (float*)(ws + (size_t)40 * 1024 * 1024 + 256 * 1024);
  float* q2c     = (float*)(ws + (size_t)40 * 1024 * 1024 + 256 * 1024 + 4096);

  k_conv_q<<<(NB * NQ * ND) / 256, 256, 0, stream>>>(q, qbf, qT);
  k_conv_p<<<(NB * NP * ND) / 256, 256, 0, stream>>>(pass, pbf);

  dim3 g1(NP / 64, NB);
  k_bidaf<<<g1, 128, 0, stream>>>(qbf, qT, pbf, pass, out, max_sim);

  k_q2c_stats<<<NB, 256, 0, stream>>>(max_sim, MZ, q2c);

  dim3 g2(NP / 256, NB);
  k_q2c_acc<<<g2, 256, 0, stream>>>(max_sim, MZ, pass, q2c);

  k_merge<<<(NB * NP * ND / 4) / 256, 256, 0, stream>>>(pass, q2c, out);
}
